// LlamaAttention_70927089926603
// MI455X (gfx1250) — compile-verified, matrix operand order checked by reading, not run on hardware
//
#include <hip/hip_runtime.h>
#include <hip/hip_bf16.h>

// ---------------------------------------------------------------------------
// CDNA5 (gfx1250) wave32 WMMA implementation of the FlatQuant Llama attention
// layer.  All matmuls (token transform, QKV, Mk rotations, attention S & PV,
// output projection) run on v_wmma_f32_16x16x32_f16 with fp32 accumulation.
// GEMM tile staging uses the CDNA5 async global->LDS path (ASYNCcnt) with
// double-buffered LDS so VMEM traffic overlaps the WMMA pipeline.
// ---------------------------------------------------------------------------

typedef __attribute__((ext_vector_type(16))) _Float16 v16h;
typedef __attribute__((ext_vector_type(8)))  _Float16 v8h;
typedef __attribute__((ext_vector_type(8)))  float    v8f;

#define T_TOK   2048
#define HID     4096
#define NHEAD   32
#define NKV     8
#define DHEAD   128
#define NQKV    (NHEAD*DHEAD + 2*NKV*DHEAD)   // 6144
#define QMAX4   7.0f
#define QMIN4   (-8.0f)
#define NEGINF  (-1e30f)

static __device__ __forceinline__ v8f wmma_f16(v16h a, v16h b, v8f c) {
  // D = A(16x32 f16) * B(32x16 f16) + C(16x16 f32)
  return __builtin_amdgcn_wmma_f32_16x16x32_f16(false, a, false, b, (short)0, c,
                                                false, false);
}

static __device__ __forceinline__ v8f v8f_zero() {
  v8f z = {0.f,0.f,0.f,0.f,0.f,0.f,0.f,0.f};
  return z;
}

// ---------------------------------------------------------------------------
// CDNA5 async global->LDS copy (16 bytes per lane, tracked by ASYNCcnt).
// VDST VGPR holds the LDS byte address (low 32 bits of the generic pointer:
// the LDS aperture maps addr[31:0] to the LDS offset, ISA 10.2).
// ---------------------------------------------------------------------------
static __device__ __forceinline__ void async_copy_b128(void* lds_ptr, const void* gptr) {
  uint32_t lds_off = (uint32_t)(uintptr_t)lds_ptr;
  uint64_t gaddr   = (uint64_t)(uintptr_t)gptr;
  asm volatile("global_load_async_to_lds_b128 %0, %1, off"
               :: "v"(lds_off), "v"(gaddr)
               : "memory");
}
static __device__ __forceinline__ void wait_async0() {
  asm volatile("s_wait_asynccnt 0x0" ::: "memory");
}
static __device__ __forceinline__ void wait_async4() {
  asm volatile("s_wait_asynccnt 0x4" ::: "memory");
}

// Load a 16x32 f16 fragment in the CDNA5 WMMA A-operand layout from row-major
// storage with leading dimension `ld` (in halves, multiple of 8).  For the B
// operand pass a pointer to B^T stored row-major (N rows of K halves): the
// per-lane addressing is identical (A-layout of X == B-layout of X^T).
//   lane l: row = l&15, kbase = (l&16)?8:0, elems j<8 -> k=kbase+j,
//           elems j>=8 -> k=kbase+16+(j-8)       (ISA 7.12.2, 16-bit A 16x32)
static __device__ __forceinline__ v16h load_frag_rm(const _Float16* p, int ld) {
  const int lane = threadIdx.x & 31;
  const _Float16* rp = p + (lane & 15) * ld + ((lane & 16) ? 8 : 0);
  v8h lo = *(const v8h*)(rp);
  v8h hi = *(const v8h*)(rp + 16);
  v16h f;
#pragma unroll
  for (int i = 0; i < 8; ++i) { f[i] = lo[i]; f[8 + i] = hi[i]; }
  return f;
}

// ---------------------------------------------------------------------------
// fp32 -> f16 transpose: out[n*K + k] = (f16) in[k*N + n]   (makes B^T copies)
// ---------------------------------------------------------------------------
__global__ __launch_bounds__(256) void transpose_cvt_kernel(
    const float* __restrict__ in, _Float16* __restrict__ out, int K, int N) {
  long gid = (long)blockIdx.x * 256 + threadIdx.x;
  if (gid >= (long)K * N) return;
  int k = (int)(gid / N), n = (int)(gid % N);
  out[(size_t)n * K + k] = (_Float16)in[gid];
}

// ---------------------------------------------------------------------------
// Kernel 1: per-token  out = Ml @ H_t @ Mr  (64x64x64 twice, WMMA) followed by
// per-token symmetric 4-bit fake-quant; writes dequantized f16 activations.
// One block (4 waves) per token.
// ---------------------------------------------------------------------------
__global__ __launch_bounds__(128) void transform_quant_kernel(
    const float* __restrict__ hs, const float* __restrict__ Ml,
    const float* __restrict__ Mr, _Float16* __restrict__ A1) {
  __shared__ __align__(16) _Float16 sMl[64 * 64];   // row-major (l,a)
  __shared__ __align__(16) _Float16 sHt[64 * 64];   // H_t^T : [b][a]
  __shared__ __align__(16) _Float16 sTmp[64 * 64];  // row-major (l,b)
  __shared__ __align__(16) _Float16 sMrT[64 * 64];  // Mr^T : [r][b]
  __shared__ float red[4];

  const int t    = blockIdx.x;
  const int tid  = threadIdx.x;
  const int wave = tid >> 5, lane = tid & 31;

  for (int i = tid; i < 64 * 64; i += 128) {
    int r = i >> 6, c = i & 63;
    sMl[i]           = (_Float16)Ml[i];
    sHt[c * 64 + r]  = (_Float16)hs[(size_t)t * HID + i];
    sMrT[c * 64 + r] = (_Float16)Mr[i];
  }
  __syncthreads();

  const int l0 = wave * 16;
  const int n  = lane & 15, mb = (lane & 16) ? 8 : 0;

  // GEMM1: tmp = Ml @ H_t
  v8f acc[4];
#pragma unroll
  for (int nt = 0; nt < 4; ++nt) acc[nt] = v8f_zero();
#pragma unroll
  for (int kc = 0; kc < 2; ++kc) {
    v16h a = load_frag_rm(sMl + l0 * 64 + kc * 32, 64);
#pragma unroll
    for (int nt = 0; nt < 4; ++nt) {
      v16h b = load_frag_rm(sHt + nt * 16 * 64 + kc * 32, 64);
      acc[nt] = wmma_f16(a, b, acc[nt]);
    }
  }
#pragma unroll
  for (int nt = 0; nt < 4; ++nt)
#pragma unroll
    for (int r = 0; r < 8; ++r)
      sTmp[(l0 + mb + r) * 64 + nt * 16 + n] = (_Float16)acc[nt][r];
  __syncthreads();

  // GEMM2: out = tmp @ Mr
  v8f acc2[4];
#pragma unroll
  for (int nt = 0; nt < 4; ++nt) acc2[nt] = v8f_zero();
#pragma unroll
  for (int kc = 0; kc < 2; ++kc) {
    v16h a = load_frag_rm(sTmp + l0 * 64 + kc * 32, 64);
#pragma unroll
    for (int nt = 0; nt < 4; ++nt) {
      v16h b = load_frag_rm(sMrT + nt * 16 * 64 + kc * 32, 64);
      acc2[nt] = wmma_f16(a, b, acc2[nt]);
    }
  }

  // per-token absmax -> scale -> fake-quant
  float am = 0.f;
#pragma unroll
  for (int nt = 0; nt < 4; ++nt)
#pragma unroll
    for (int r = 0; r < 8; ++r) am = fmaxf(am, fabsf(acc2[nt][r]));
#pragma unroll
  for (int off = 16; off >= 1; off >>= 1) am = fmaxf(am, __shfl_xor(am, off));
  if (lane == 0) red[wave] = am;
  __syncthreads();
  float gm  = fmaxf(fmaxf(red[0], red[1]), fmaxf(red[2], red[3]));
  float s   = fmaxf(gm * (1.f / QMAX4), 1e-8f);
  float inv = 1.f / s;
#pragma unroll
  for (int nt = 0; nt < 4; ++nt)
#pragma unroll
    for (int r = 0; r < 8; ++r) {
      float q = rintf(acc2[nt][r] * inv);
      q = fminf(fmaxf(q, QMIN4), QMAX4);
      A1[(size_t)t * HID + (l0 + mb + r) * 64 + nt * 16 + n] = (_Float16)(q * s);
    }
}

// ---------------------------------------------------------------------------
// Generic WMMA GEMM with async double-buffered LDS staging:
//   C[MxN](OutT) = A[MxK](f16,rm) * B[KxN],  B given transposed (Bt: N x K).
// Block = 256 thr (8 waves), tile 128x128, BK=32; each wave computes 32x64.
// Slab k+1 streams global->LDS via ASYNCcnt while slab k runs on the WMMAs.
// M,N multiples of 128; K multiple of 32.
// ---------------------------------------------------------------------------
template <typename OutT>
__global__ __launch_bounds__(256) void gemm_wmma_kernel(
    const _Float16* __restrict__ A, const _Float16* __restrict__ Bt,
    OutT* __restrict__ C, int M, int N, int K) {
  __shared__ __align__(16) _Float16 sA[2][128 * 32];
  __shared__ __align__(16) _Float16 sB[2][128 * 32];
  const int m0 = blockIdx.y * 128, n0 = blockIdx.x * 128;
  const int tid = threadIdx.x;
  const int wave = tid >> 5, lane = tid & 31;
  const int wm = wave & 3, wn = wave >> 2;  // 4 row-groups x 2 col-groups

  // per-thread staging chunks (2 x v8h per operand per slab)
  int srow[2], sc8[2];
#pragma unroll
  for (int c = 0; c < 2; ++c) {
    int cc = tid + c * 256;
    srow[c] = cc >> 2;
    sc8[c]  = (cc & 3) * 8;
  }

  auto prefetch = [&](int buf, int k0) {
#pragma unroll
    for (int c = 0; c < 2; ++c) {
      async_copy_b128(&sA[buf][srow[c] * 32 + sc8[c]],
                      A + (size_t)(m0 + srow[c]) * K + k0 + sc8[c]);
      async_copy_b128(&sB[buf][srow[c] * 32 + sc8[c]],
                      Bt + (size_t)(n0 + srow[c]) * K + k0 + sc8[c]);
    }
  };

  v8f acc[2][4];
#pragma unroll
  for (int mt = 0; mt < 2; ++mt)
#pragma unroll
    for (int nt = 0; nt < 4; ++nt) acc[mt][nt] = v8f_zero();

  prefetch(0, 0);
  int cur = 0;
  for (int k0 = 0; k0 < K; k0 += 32) {
    if (k0 + 32 < K) {
      prefetch(cur ^ 1, k0 + 32);   // stream next slab while this one computes
      wait_async4();                // oldest 4 (this slab) complete, in order
    } else {
      wait_async0();
    }
    __syncthreads();
    v16h af[2];
#pragma unroll
    for (int mt = 0; mt < 2; ++mt)
      af[mt] = load_frag_rm(&sA[cur][(wm * 32 + mt * 16) * 32], 32);
#pragma unroll
    for (int nt = 0; nt < 4; ++nt) {
      v16h bf = load_frag_rm(&sB[cur][(wn * 64 + nt * 16) * 32], 32);
#pragma unroll
      for (int mt = 0; mt < 2; ++mt) acc[mt][nt] = wmma_f16(af[mt], bf, acc[mt][nt]);
    }
    __syncthreads();
    cur ^= 1;
  }

  const int n = lane & 15, mb = (lane & 16) ? 8 : 0;
#pragma unroll
  for (int mt = 0; mt < 2; ++mt)
#pragma unroll
    for (int nt = 0; nt < 4; ++nt)
#pragma unroll
      for (int r = 0; r < 8; ++r)
        C[(size_t)(m0 + wm * 32 + mt * 16 + mb + r) * N +
          (n0 + wn * 64 + nt * 16 + n)] = (OutT)acc[mt][nt][r];
}

// ---------------------------------------------------------------------------
// RoPE (NeoX) on Q and K, fp32 -> f16.  One thread per (row, i<64).
// ---------------------------------------------------------------------------
__global__ __launch_bounds__(256) void rope_kernel(
    const float* __restrict__ qkv, const int* __restrict__ pos,
    _Float16* __restrict__ Qr, _Float16* __restrict__ Kr) {
  int gid = blockIdx.x * 256 + threadIdx.x;
  const int total = T_TOK * (NHEAD + NKV) * 64;
  if (gid >= total) return;
  int i = gid & 63, row = gid >> 6;
  int t = row / (NHEAD + NKV), hh = row % (NHEAD + NKV);
  float p    = (float)pos[t];
  float freq = __powf(10000.0f, -(float)(2 * i) * (1.0f / (float)DHEAD));
  float ang  = p * freq;
  float sn, cs;
  __sincosf(ang, &sn, &cs);
  const float* src;
  _Float16* dst;
  if (hh < NHEAD) {
    src = qkv + (size_t)t * NQKV + hh * DHEAD;
    dst = Qr + ((size_t)t * NHEAD + hh) * DHEAD;
  } else {
    int kvh = hh - NHEAD;
    src = qkv + (size_t)t * NQKV + NHEAD * DHEAD + kvh * DHEAD;
    dst = Kr + ((size_t)t * NKV + kvh) * DHEAD;
  }
  float x1 = src[i], x2 = src[i + 64];
  dst[i]      = (_Float16)(x1 * cs - x2 * sn);
  dst[i + 64] = (_Float16)(x2 * cs + x1 * sn);
}

// ---------------------------------------------------------------------------
// V: per-row (group=128) 4-bit fake-quant, fp32 -> f16.  One wave per row.
// ---------------------------------------------------------------------------
__global__ __launch_bounds__(256) void vquant_kernel(
    const float* __restrict__ qkv, _Float16* __restrict__ Vq) {
  const int tid = threadIdx.x, wave = tid >> 5, lane = tid & 31;
  const int row = blockIdx.x * 8 + wave;      // 0 .. T*NKV-1
  const int t = row >> 3, kvh = row & 7;
  const float* src = qkv + (size_t)t * NQKV + (NHEAD + NKV) * DHEAD + kvh * DHEAD;
  float v[4];
  float am = 0.f;
#pragma unroll
  for (int j = 0; j < 4; ++j) {
    v[j] = src[lane + j * 32];
    am = fmaxf(am, fabsf(v[j]));
  }
#pragma unroll
  for (int off = 16; off >= 1; off >>= 1) am = fmaxf(am, __shfl_xor(am, off));
  float s = fmaxf(am * (1.f / QMAX4), 1e-8f), inv = 1.f / s;
#pragma unroll
  for (int j = 0; j < 4; ++j) {
    float q = rintf(v[j] * inv);
    q = fminf(fmaxf(q, QMIN4), QMAX4);
    Vq[(size_t)row * DHEAD + lane + j * 32] = (_Float16)(q * s);
  }
}

// ---------------------------------------------------------------------------
// Mk rotation GEMM: Out[Mx128] = A[Mx128] @ Mk (MkT async-staged in LDS), f16.
// Mk orthogonal  =>  inv(Mk)^T == Mk, so Q and K share MkT.
// Block = 128 thr (4 waves), each wave 16 rows x 128 cols (32 WMMA).
// QUANT: fused per-row (group=128) 4-bit fake-quant (K path).
// ---------------------------------------------------------------------------
template <bool QUANT>
__global__ __launch_bounds__(128) void gemm_mk_kernel(
    const _Float16* __restrict__ A, const _Float16* __restrict__ MkT,
    _Float16* __restrict__ Out) {
  __shared__ __align__(16) _Float16 sB[128 * 128];
  const int tid = threadIdx.x, wave = tid >> 5, lane = tid & 31;
  for (int c = tid; c < 2048; c += 128) {
    int row = c >> 4, c8 = (c & 15) * 8;
    async_copy_b128(sB + row * 128 + c8, MkT + row * 128 + c8);
  }
  wait_async0();
  __syncthreads();

  const int row0 = blockIdx.x * 64 + wave * 16;
  v8f acc[8];
#pragma unroll
  for (int nt = 0; nt < 8; ++nt) acc[nt] = v8f_zero();
#pragma unroll
  for (int kc = 0; kc < 4; ++kc) {
    v16h a = load_frag_rm(A + (size_t)row0 * DHEAD + kc * 32, DHEAD);
#pragma unroll
    for (int nt = 0; nt < 8; ++nt) {
      v16h b = load_frag_rm(sB + nt * 16 * 128 + kc * 32, 128);
      acc[nt] = wmma_f16(a, b, acc[nt]);
    }
  }

  const int n = lane & 15, mb = (lane & 16) ? 8 : 0;
  if (QUANT) {
    float rm[8];
#pragma unroll
    for (int r = 0; r < 8; ++r) {
      float m = 0.f;
#pragma unroll
      for (int nt = 0; nt < 8; ++nt) m = fmaxf(m, fabsf(acc[nt][r]));
#pragma unroll
      for (int off = 8; off >= 1; off >>= 1) m = fmaxf(m, __shfl_xor(m, off));
      rm[r] = m;
    }
#pragma unroll
    for (int r = 0; r < 8; ++r) {
      float s = fmaxf(rm[r] * (1.f / QMAX4), 1e-8f), inv = 1.f / s;
#pragma unroll
      for (int nt = 0; nt < 8; ++nt) {
        float q = rintf(acc[nt][r] * inv);
        q = fminf(fmaxf(q, QMIN4), QMAX4);
        Out[(size_t)(row0 + mb + r) * DHEAD + nt * 16 + n] = (_Float16)(q * s);
      }
    }
  } else {
#pragma unroll
    for (int nt = 0; nt < 8; ++nt)
#pragma unroll
      for (int r = 0; r < 8; ++r)
        Out[(size_t)(row0 + mb + r) * DHEAD + nt * 16 + n] = (_Float16)acc[nt][r];
  }
}

// ---------------------------------------------------------------------------
// Causal GQA flash attention.  Grid = (T/64)*H blocks, 128 thr (4 waves,
// 16 queries each).  Per 64-key block: K block async-staged in LDS, V^T
// staged with a VALU transpose, S = Q@K^T via WMMA, online softmax (width-16
// shfl reductions matching the C/D layout), P staged through per-wave LDS to
// reach the A-layout, O += P@V via WMMA.
// ---------------------------------------------------------------------------
__global__ __launch_bounds__(128) void attn_kernel(
    const _Float16* __restrict__ Q,   // (T, H, D) f16
    const _Float16* __restrict__ Kq,  // (T, KV, D) f16
    const _Float16* __restrict__ Vq,  // (T, KV, D) f16
    float* __restrict__ O) {          // (T, H, D) f32
  __shared__ __align__(16) _Float16 sK[64 * 128];    // [key][d]
  __shared__ __align__(16) _Float16 sVt[128 * 64];   // [d][key]
  __shared__ __align__(16) _Float16 sP[4][16 * 64];  // per-wave [qrow][key]

  const int h  = blockIdx.x & (NHEAD - 1);
  const int qb = blockIdx.x >> 5;
  const int kvh = h >> 2;                            // rep = H/KV = 4
  const int tid = threadIdx.x, wave = tid >> 5, lane = tid & 31;
  const int q0 = qb * 64 + wave * 16;
  const int n = lane & 15, mb = (lane & 16) ? 8 : 0;
  const float scale = 0.08838834764831845f;          // D^-0.5

  v16h qa[4];
  const _Float16* Qbase = Q + ((size_t)q0 * NHEAD + h) * DHEAD;
#pragma unroll
  for (int kc = 0; kc < 4; ++kc) qa[kc] = load_frag_rm(Qbase + kc * 32, NHEAD * DHEAD);

  v8f o[8];
#pragma unroll
  for (int d = 0; d < 8; ++d) o[d] = v8f_zero();
  float row_m[8], row_l[8];
#pragma unroll
  for (int r = 0; r < 8; ++r) { row_m[r] = -3.0e38f; row_l[r] = 0.f; }

  const int nblocks = qb + 1;                        // causal block bound
  for (int jb = 0; jb < nblocks; ++jb) {
    const int j0 = jb * 64;
    __syncthreads();
    // stage K (async, no VGPR round-trip) and V^T (VALU transpose)
    for (int c = tid; c < 1024; c += 128) {
      int row = c >> 4, c8 = (c & 15) * 8;
      const size_t src = ((size_t)(j0 + row) * NKV + kvh) * DHEAD + c8;
      async_copy_b128(sK + row * 128 + c8, Kq + src);
      v8h vv = *(const v8h*)(Vq + src);
#pragma unroll
      for (int i = 0; i < 8; ++i) sVt[(c8 + i) * 64 + row] = vv[i];
    }
    wait_async0();
    __syncthreads();

    // S = Q @ K^T  (16x64 per wave)
    v8f sacc[4];
#pragma unroll
    for (int nt = 0; nt < 4; ++nt) sacc[nt] = v8f_zero();
#pragma unroll
    for (int kc = 0; kc < 4; ++kc)
#pragma unroll
      for (int nt = 0; nt < 4; ++nt) {
        v16h bf = load_frag_rm(sK + nt * 16 * 128 + kc * 32, 128);
        sacc[nt] = wmma_f16(qa[kc], bf, sacc[nt]);
      }

    // scale + causal mask + online softmax
    float pl[4][8], cm[8];
#pragma unroll
    for (int r = 0; r < 8; ++r) cm[r] = -3.0e38f;
#pragma unroll
    for (int nt = 0; nt < 4; ++nt)
#pragma unroll
      for (int r = 0; r < 8; ++r) {
        float v = sacc[nt][r] * scale;
        if (j0 + nt * 16 + n > q0 + mb + r) v = NEGINF;
        pl[nt][r] = v;
        cm[r] = fmaxf(cm[r], v);
      }
#pragma unroll
    for (int r = 0; r < 8; ++r) {
#pragma unroll
      for (int off = 8; off >= 1; off >>= 1) cm[r] = fmaxf(cm[r], __shfl_xor(cm[r], off));
      float nm    = fmaxf(row_m[r], cm[r]);
      float alpha = __expf(row_m[r] - nm);
      row_m[r]    = nm;
      float ls = 0.f;
#pragma unroll
      for (int nt = 0; nt < 4; ++nt) {
        float p = __expf(pl[nt][r] - nm);
        pl[nt][r] = p;
        ls += p;
      }
#pragma unroll
      for (int off = 8; off >= 1; off >>= 1) ls += __shfl_xor(ls, off);
      row_l[r] = row_l[r] * alpha + ls;
#pragma unroll
      for (int d = 0; d < 8; ++d) o[d][r] *= alpha;
    }

    // P -> per-wave LDS (D-layout -> row-major) for A-layout reload
    _Float16* pw = &sP[wave][0];
#pragma unroll
    for (int nt = 0; nt < 4; ++nt)
#pragma unroll
      for (int r = 0; r < 8; ++r)
        pw[(mb + r) * 64 + nt * 16 + n] = (_Float16)pl[nt][r];
    __syncthreads();

    // O += P @ V
    v16h pa[2];
#pragma unroll
    for (int kc = 0; kc < 2; ++kc) pa[kc] = load_frag_rm(pw + kc * 32, 64);
#pragma unroll
    for (int d = 0; d < 8; ++d)
#pragma unroll
      for (int kc = 0; kc < 2; ++kc) {
        v16h vb = load_frag_rm(sVt + d * 16 * 64 + kc * 32, 64);
        o[d] = wmma_f16(pa[kc], vb, o[d]);
      }
  }

  // epilogue: normalize and store fp32 (T,H,D)
#pragma unroll
  for (int d = 0; d < 8; ++d)
#pragma unroll
    for (int r = 0; r < 8; ++r)
      O[((size_t)(q0 + mb + r) * NHEAD + h) * DHEAD + d * 16 + n] =
          o[d][r] * (1.f / row_l[r]);
}

// ---------------------------------------------------------------------------
// Head mixing (out'[t,h,d] = sum_g Mo[g,h] * out[t,g,d]) + per-token 4-bit
// fake-quant.  One block (256 thr) per token; 0.27 GFLOP total => VALU is fine.
// ---------------------------------------------------------------------------
__global__ __launch_bounds__(256) void momix_quant_kernel(
    const float* __restrict__ X, const float* __restrict__ Mo,
    _Float16* __restrict__ A2) {
  __shared__ float sX[NHEAD * DHEAD];
  __shared__ float sMo[NHEAD * NHEAD];
  __shared__ float red[8];
  const int t = blockIdx.x, tid = threadIdx.x;
  const int wave = tid >> 5, lane = tid & 31;
  for (int i = tid; i < NHEAD * DHEAD; i += 256) sX[i] = X[(size_t)t * HID + i];
  for (int i = tid; i < NHEAD * NHEAD; i += 256) sMo[i] = Mo[i];
  __syncthreads();

  float val[16];
  float am = 0.f;
#pragma unroll
  for (int j = 0; j < 16; ++j) {
    int idx = tid + j * 256;
    int hh = idx >> 7, d = idx & 127;
    float acc = 0.f;
#pragma unroll
    for (int g = 0; g < NHEAD; ++g) acc += sMo[g * NHEAD + hh] * sX[g * DHEAD + d];
    val[j] = acc;
    am = fmaxf(am, fabsf(acc));
  }
#pragma unroll
  for (int off = 16; off >= 1; off >>= 1) am = fmaxf(am, __shfl_xor(am, off));
  if (lane == 0) red[wave] = am;
  __syncthreads();
  float gm = 0.f;
#pragma unroll
  for (int w = 0; w < 8; ++w) gm = fmaxf(gm, red[w]);
  float s = fmaxf(gm * (1.f / QMAX4), 1e-8f), inv = 1.f / s;
#pragma unroll
  for (int j = 0; j < 16; ++j) {
    int idx = tid + j * 256;
    float q = rintf(val[j] * inv);
    q = fminf(fmaxf(q, QMIN4), QMAX4);
    A2[(size_t)t * HID + idx] = (_Float16)(q * s);
  }
}

// ---------------------------------------------------------------------------
// Host launcher
// ---------------------------------------------------------------------------
extern "C" void kernel_launch(void* const* d_in, const int* in_sizes, int n_in,
                              void* d_out, int out_size, void* d_ws, size_t ws_size,
                              hipStream_t stream) {
  (void)in_sizes; (void)out_size; (void)ws_size;
  if (n_in < 8) return;
  const int*   positions = (const int*)d_in[0];
  const float* hs        = (const float*)d_in[1];
  const float* Wqkv      = (const float*)d_in[2];
  const float* Wo        = (const float*)d_in[3];
  const float* Ml        = (const float*)d_in[4];
  const float* Mr        = (const float*)d_in[5];
  const float* Mo        = (const float*)d_in[6];
  const float* Mk        = (const float*)d_in[7];
  float* out = (float*)d_out;

  char* ws = (char*)d_ws;
  size_t off = 0;
  auto take = [&](size_t bytes) -> void* {
    void* p = ws + off;
    off += (bytes + 255) & ~(size_t)255;
    return p;
  };
  _Float16* WqkvT = (_Float16*)take((size_t)NQKV * HID * 2);          // 48 MiB
  _Float16* WoT   = (_Float16*)take((size_t)HID * HID * 2);           // 32 MiB
  _Float16* MkT   = (_Float16*)take((size_t)DHEAD * DHEAD * 2);
  _Float16* A1    = (_Float16*)take((size_t)T_TOK * HID * 2);         // 16 MiB
  float*    QKVo  = (float*)take((size_t)T_TOK * NQKV * 4);           // 48 MiB
  _Float16* Qr    = (_Float16*)take((size_t)T_TOK * NHEAD * DHEAD * 2);
  _Float16* Kr    = (_Float16*)take((size_t)T_TOK * NKV * DHEAD * 2);
  _Float16* Q2    = (_Float16*)take((size_t)T_TOK * NHEAD * DHEAD * 2);
  _Float16* Kq    = (_Float16*)take((size_t)T_TOK * NKV * DHEAD * 2);
  _Float16* Vq    = (_Float16*)take((size_t)T_TOK * NKV * DHEAD * 2);
  float*    Ao    = (float*)take((size_t)T_TOK * HID * 4);            // 32 MiB
  _Float16* A2    = (_Float16*)take((size_t)T_TOK * HID * 2);

  // weight f16 transposes (B^T copies for contiguous B-fragment loads)
  transpose_cvt_kernel<<<((size_t)HID * NQKV) / 256, 256, 0, stream>>>(Wqkv, WqkvT, HID, NQKV);
  transpose_cvt_kernel<<<((size_t)HID * HID) / 256, 256, 0, stream>>>(Wo, WoT, HID, HID);
  transpose_cvt_kernel<<<(DHEAD * DHEAD) / 256, 256, 0, stream>>>(Mk, MkT, DHEAD, DHEAD);

  // 1) per-token Kronecker transform + 4-bit activation fake-quant
  transform_quant_kernel<<<T_TOK, 128, 0, stream>>>(hs, Ml, Mr, A1);

  // 2) QKV projection: (2048x4096) @ (4096x6144)
  gemm_wmma_kernel<float><<<dim3(NQKV / 128, T_TOK / 128), 256, 0, stream>>>(
      A1, WqkvT, QKVo, T_TOK, NQKV, HID);

  // 3) RoPE on Q/K; V per-row 4-bit quant
  rope_kernel<<<(T_TOK * (NHEAD + NKV) * 64) / 256, 256, 0, stream>>>(QKVo, positions, Qr, Kr);
  vquant_kernel<<<(T_TOK * NKV) / 8, 256, 0, stream>>>(QKVo, Vq);

  // 4) Mk rotations (inv(Mk)^T == Mk for orthogonal Mk); K fused-quantized
  gemm_mk_kernel<false><<<(T_TOK * NHEAD) / 64, 128, 0, stream>>>(Qr, MkT, Q2);
  gemm_mk_kernel<true><<<(T_TOK * NKV) / 64, 128, 0, stream>>>(Kr, MkT, Kq);

  // 5) causal GQA flash attention
  attn_kernel<<<(T_TOK / 64) * NHEAD, 128, 0, stream>>>(Q2, Kq, Vq, Ao);

  // 6) head mixing (Mo) + per-token 4-bit fake-quant
  momix_quant_kernel<<<T_TOK, 256, 0, stream>>>(Ao, Mo, A2);

  // 7) output projection: (2048x4096) @ (4096x4096) -> fp32 d_out
  gemm_wmma_kernel<float><<<dim3(HID / 128, T_TOK / 128), 256, 0, stream>>>(
      A2, WoT, out, T_TOK, HID, HID);
}
